// RGTSR_21079699489029
// MI455X (gfx1250) — matrix operand order repaired and checked by reading, not compiled, not run on hardware
//
#include <hip/hip_runtime.h>
#include <stdint.h>

#define NN   200000
#define DD   64
#define QN   128
#define EE   1024
#define NEE  (QN*EE)    // 131072 edges
#define KSEL 128
#define PSEL (QN*KSEL)  // 16384 selected edges

typedef __attribute__((ext_vector_type(16))) __bf16 v16bf;
typedef __attribute__((ext_vector_type(8)))  float  v8f;
typedef __attribute__((ext_vector_type(2)))  float  v2f;

union F16x16 { uint32_t u[8]; v16bf v; };
union C8     { float f[8];  v8f v; };
union C2     { float f[2];  v2f v; };

__device__ __forceinline__ uint16_t f2bf(float f) {
  uint32_t u = __float_as_uint(f);
  u += 0x7FFFu + ((u >> 16) & 1u);   // round-to-nearest-even
  return (uint16_t)(u >> 16);
}
// monotonic uint key for float atomicMax
__device__ __forceinline__ uint32_t fkey(float f) {
  uint32_t u = __float_as_uint(f);
  return (u & 0x80000000u) ? ~u : (u | 0x80000000u);
}
__device__ __forceinline__ float unkey(uint32_t k) {
  uint32_t u = (k & 0x80000000u) ? (k & 0x7FFFFFFFu) : ~k;
  return __uint_as_float(u);
}

// ---------------- M = Wq^T @ Wk (256x256), plus bf16 copy of M[:,0:128] -----
__global__ __launch_bounds__(256) void k_buildM(const float* __restrict__ Wq,
                                                const float* __restrict__ Wk,
                                                float* __restrict__ M,
                                                uint16_t* __restrict__ Abf) {
  const int i = blockIdx.x, j = threadIdx.x;
  float acc = 0.f;
  for (int o = 0; o < 256; ++o)
    acc = fmaf(Wq[o*256 + i], Wk[o*256 + j], acc);
  M[i*256 + j] = acc;
  if (j < 128) Abf[i*128 + j] = f2bf(acc);
}

// ---------------- per-query bias g[q] = M[:,128:192]@qs + M[:,192:256]@qr ---
__global__ __launch_bounds__(256) void k_gq(const float* __restrict__ M,
                                            const float* __restrict__ qse,
                                            const float* __restrict__ qre,
                                            float* __restrict__ gq) {
  const int q = blockIdx.x, m = threadIdx.x;
  const float* Mr = M + (size_t)m*256;
  float acc = 0.f;
  for (int k = 0; k < 64; ++k) {
    acc = fmaf(Mr[128+k], qse[q*64+k], acc);
    acc = fmaf(Mr[192+k], qre[q*64+k], acc);
  }
  gq[q*256 + m] = acc;
}

// ---------------- zero-init accumulators -----------------------------------
__global__ void k_fill(uint32_t* __restrict__ segmax, float* __restrict__ segsum,
                       int* __restrict__ flags, float* __restrict__ outscore) {
  int i = blockIdx.x*blockDim.x + threadIdx.x;
  if (i < NN) { segmax[i] = 0u; segsum[i] = 0.f; flags[i] = 0; outscore[i] = 0.f; }
}

// ---------------- edge logits: bf16 WMMA ------------------------------------
// One block = 16 edges. 4 waves; wave wv computes w rows [wv*64, wv*64+64).
// w = M[:,0:128] @ [n_dst; rel]  (+ per-query bias as C init), logit = l . w
__global__ __launch_bounds__(128) void k_edge(
    const float* __restrict__ node_repr, const float* __restrict__ rel_emb,
    const float* __restrict__ qse, const float* __restrict__ qre,
    const int* __restrict__ edges, const uint16_t* __restrict__ Abf,
    const float* __restrict__ gq, float* __restrict__ logits)
{
  __shared__ __align__(16) uint16_t Gl[16][128];  // [edge n][k] bf16: [n_dst(64); rel(64)]
  __shared__ float Ll[16][256];                    // [edge n][k] f32:  [n_src; rel; qs; qr]
  __shared__ float lsum[16];

  const int tid = threadIdx.x;
  const int e0  = blockIdx.x << 4;
  const int q   = edges[(size_t)e0*8 + 0];         // same for all 16 edges (E%16==0)

  { // gather phase: thread -> edge n = tid>>3, segment seg = tid&7
    const int n = tid >> 3, seg = tid & 7;
    const int e = e0 + n;
    const int src = edges[(size_t)e*8 + 6];
    const int dst = edges[(size_t)e*8 + 7];
    { // right side -> bf16 LDS, 16 floats per thread
      const int kk = seg * 16;
      const float* base = (seg < 4) ? (node_repr + (size_t)dst*DD + kk)
                                    : (rel_emb  + (size_t)e*DD + (kk - 64));
      for (int i = 0; i < 16; i += 4) {
        float4 f = *(const float4*)(base + i);
        Gl[n][kk+i+0] = f2bf(f.x); Gl[n][kk+i+1] = f2bf(f.y);
        Gl[n][kk+i+2] = f2bf(f.z); Gl[n][kk+i+3] = f2bf(f.w);
      }
    }
    { // left side -> f32 LDS, 32 floats per thread
      const int kk = seg * 32;
      const float* base;
      if      (kk <  64) base = node_repr + (size_t)src*DD + kk;
      else if (kk < 128) base = rel_emb   + (size_t)e*DD + (kk - 64);
      else if (kk < 192) base = qse       + (size_t)q*DD + (kk - 128);
      else               base = qre       + (size_t)q*DD + (kk - 192);
      for (int i = 0; i < 32; i += 4) {
        float4 f = *(const float4*)(base + i);
        Ll[n][kk+i+0]=f.x; Ll[n][kk+i+1]=f.y; Ll[n][kk+i+2]=f.z; Ll[n][kk+i+3]=f.w;
      }
    }
  }
  if (tid < 16) lsum[tid] = 0.f;
  __syncthreads();

  const int wv    = tid >> 5;
  const int lane  = tid & 31;
  const int half  = lane >> 4;
  const int nn16  = lane & 15;
  const int mbase = wv * 64;

  C8 c[4];
  #pragma unroll
  for (int mt = 0; mt < 4; ++mt)
    #pragma unroll
    for (int v = 0; v < 8; ++v)
      c[mt].f[v] = gq[(size_t)q*256 + mbase + mt*16 + v + 8*half];

  for (int s = 0; s < 4; ++s) {              // K = 128, 32 per WMMA
    const int k0 = s * 32;
    // B fragment (32x16 bf16): lane n=nn16; VGPR v holds K = half*16 + 2v, +1
    F16x16 bfrag;
    const uint32_t* grow = (const uint32_t*)(&Gl[nn16][0]);
    const int kb = (k0 + half*16) >> 1;
    #pragma unroll
    for (int v = 0; v < 8; ++v) bfrag.u[v] = grow[kb + v];
    #pragma unroll
    for (int mt = 0; mt < 4; ++mt) {
      // A fragment (16x32 bf16): lane m=nn16; VGPR v holds K = half*16 + 2v, +1
      const int m = mbase + mt*16 + nn16;
      const uint32_t* arow = (const uint32_t*)(Abf + (size_t)m*128);
      F16x16 afrag;
      const int ka0 = (k0 + half*16) >> 1;
      const int ka1 = ka0 + 4;
      #pragma unroll
      for (int v = 0; v < 4; ++v) { afrag.u[v] = arow[ka0 + v]; afrag.u[4+v] = arow[ka1 + v]; }
      c[mt].v = __builtin_amdgcn_wmma_f32_16x16x32_bf16(false, afrag.v, false, bfrag.v,
                                                        (short)0, c[mt].v, false, false);
    }
  }

  // logit partial: lane holds w[m][n] for m = mbase + mt*16 + v + 8*half, n = nn16
  float part = 0.f;
  #pragma unroll
  for (int mt = 0; mt < 4; ++mt)
    #pragma unroll
    for (int v = 0; v < 8; ++v)
      part += Ll[nn16][mbase + mt*16 + v + 8*half] * c[mt].f[v];
  atomicAdd(&lsum[nn16], part);
  __syncthreads();
  if (tid < 16) logits[(size_t)e0 + tid] = lsum[tid];
}

// ---------------- segment softmax over src ---------------------------------
__global__ void k_segmax(const float* __restrict__ logits, const int* __restrict__ edges,
                         uint32_t* __restrict__ segmax) {
  int i = blockIdx.x*blockDim.x + threadIdx.x;
  if (i < NEE) {
    int src = edges[(size_t)i*8 + 6];
    atomicMax(&segmax[src], fkey(logits[i]));
  }
}
__global__ void k_expsum(const float* __restrict__ logits, const int* __restrict__ edges,
                         const uint32_t* __restrict__ segmax, float* __restrict__ exbuf,
                         float* __restrict__ segsum) {
  int i = blockIdx.x*blockDim.x + threadIdx.x;
  if (i < NEE) {
    int src = edges[(size_t)i*8 + 6];
    float ex = expf(logits[i] - unkey(segmax[src]));
    exbuf[i] = ex;
    atomicAdd(&segsum[src], ex);
  }
}
__global__ void k_soft(const float* __restrict__ exbuf, const int* __restrict__ edges,
                       const float* __restrict__ segsum, const float* __restrict__ node_score,
                       float* __restrict__ softb, float* __restrict__ tscore) {
  int i = blockIdx.x*blockDim.x + threadIdx.x;
  if (i < NEE) {
    int src = edges[(size_t)i*8 + 6];
    float s = exbuf[i] / segsum[src];
    softb[i]  = s;
    tscore[i] = s * node_score[src];
  }
}

// ---------------- per-query top-128 of 1024 (bitonic, desc, low-idx ties) ---
__global__ __launch_bounds__(512) void k_topk(const float* __restrict__ tscore,
                                              int* __restrict__ sel) {
  __shared__ float v[1024];
  __shared__ int  ix[1024];
  const int q = blockIdx.x, tid = threadIdx.x;
  for (int i = tid; i < 1024; i += 512) { v[i] = tscore[(size_t)q*1024 + i]; ix[i] = i; }
  __syncthreads();
  for (int k = 2; k <= 1024; k <<= 1) {
    for (int j = k >> 1; j > 0; j >>= 1) {
      for (int t = tid; t < 1024; t += 512) {
        int p = t ^ j;
        if (p > t) {
          float a = v[t], b = v[p];
          int ia = ix[t], ib = ix[p];
          bool agtb = (a > b) || (a == b && ia < ib);
          bool desc = ((t & k) == 0);
          if (desc ? !agtb : agtb) { v[t]=b; v[p]=a; ix[t]=ib; ix[p]=ia; }
        }
      }
      __syncthreads();
    }
  }
  if (tid < KSEL) sel[q*KSEL + tid] = q*1024 + ix[tid];
}

// ---------------- scatter: node scores + src flags --------------------------
__global__ void k_scatter1(const int* __restrict__ sel, const int* __restrict__ edges,
                           const float* __restrict__ softb, const float* __restrict__ node_score,
                           float* __restrict__ outscore, int* __restrict__ flags) {
  int i = blockIdx.x*blockDim.x + threadIdx.x;
  if (i < PSEL) {
    int e = sel[i];
    int src = edges[(size_t)e*8 + 6], dst = edges[(size_t)e*8 + 7];
    atomicAdd(&outscore[dst], softb[e] * node_score[src]);
    flags[src] = 1;
  }
}

// ---------------- upd[n] = (is_src[n] ? 0 : node_repr[n]) -------------------
__global__ void k_initupd(const float* __restrict__ node_repr, const int* __restrict__ flags,
                          float* __restrict__ upd) {
  int i = blockIdx.x*blockDim.x + threadIdx.x;   // float4 granularity
  if (i < NN*16) {
    float4 x = ((const float4*)node_repr)[i];
    if (flags[i >> 4]) x = make_float4(0.f, 0.f, 0.f, 0.f);
    ((float4*)upd)[i] = x;
  }
}

// ---------------- upd[src] += soft * node_repr[dst] -------------------------
__global__ __launch_bounds__(64) void k_scatter2(const int* __restrict__ sel,
                                                 const int* __restrict__ edges,
                                                 const float* __restrict__ softb,
                                                 const float* __restrict__ node_repr,
                                                 float* __restrict__ upd) {
  const int e = sel[blockIdx.x], c = threadIdx.x;
  const int src = edges[(size_t)e*8 + 6], dst = edges[(size_t)e*8 + 7];
  atomicAdd(&upd[(size_t)src*DD + c], softb[e] * node_repr[(size_t)dst*DD + c]);
}

// ---------------- out = leaky_relu(upd @ W_lin + b), f32 WMMA, in place -----
__global__ __launch_bounds__(128) void k_out(float* __restrict__ upd,
                                             const float* __restrict__ W_lin,
                                             const float* __restrict__ b_lin) {
  const int wv = threadIdx.x >> 5, lane = threadIdx.x & 31;
  const int half = lane >> 4, nn16 = lane & 15;
  const int row0 = blockIdx.x * 64 + wv * 16;      // 16 rows per wave, disjoint
#if __has_builtin(__builtin_amdgcn_wmma_f32_16x16x4_f32)
  C8 c[4];
  #pragma unroll
  for (int nt = 0; nt < 4; ++nt)
    #pragma unroll
    for (int v = 0; v < 8; ++v) c[nt].f[v] = 0.f;
  for (int s = 0; s < 16; ++s) {                   // K = 64, 4 per WMMA
    const int k0 = s * 4;
    C2 a;                                          // A 16x4 f32: VGPR v: K = k0+half*2+v
    const float* ap = upd + (size_t)(row0 + nn16)*DD + k0 + half*2;
    a.f[0] = ap[0]; a.f[1] = ap[1];
    #pragma unroll
    for (int nt = 0; nt < 4; ++nt) {
      C2 b;                                        // B 4x16 f32: lane n; VGPR v: K = half*2+v
      const int col = nt*16 + nn16;
      b.f[0] = W_lin[(size_t)(k0 + half*2    )*DD + col];
      b.f[1] = W_lin[(size_t)(k0 + half*2 + 1)*DD + col];
      c[nt].v = __builtin_amdgcn_wmma_f32_16x16x4_f32(false, a.v, false, b.v,
                                                      (short)0, c[nt].v, false, false);
    }
  }
  #pragma unroll
  for (int nt = 0; nt < 4; ++nt) {
    const int col = nt*16 + nn16;
    const float bias = b_lin[col];
    #pragma unroll
    for (int v = 0; v < 8; ++v) {
      float val = c[nt].f[v] + bias;
      val = val > 0.f ? val : 0.01f * val;
      upd[(size_t)(row0 + v + 8*half)*DD + col] = val;  // row M = v + 8*half
    }
  }
#else
  // VALU fallback: block covers rows [blockIdx.x*64, +64), read-then-write
  float acc[32];
  const int base = blockIdx.x * 64;
  for (int t = threadIdx.x, s = 0; t < 64*64; t += 128, ++s) {
    const int r = t >> 6, j = t & 63;
    float a = b_lin[j];
    for (int k = 0; k < 64; ++k)
      a = fmaf(upd[(size_t)(base + r)*DD + k], W_lin[(size_t)k*DD + j], a);
    acc[s] = a > 0.f ? a : 0.01f * a;
  }
  __syncthreads();
  for (int t = threadIdx.x, s = 0; t < 64*64; t += 128, ++s)
    upd[(size_t)(base + (t >> 6))*DD + (t & 63)] = acc[s];
#endif
}

extern "C" void kernel_launch(void* const* d_in, const int* in_sizes, int n_in,
                              void* d_out, int out_size, void* d_ws, size_t ws_size,
                              hipStream_t stream) {
  const float* node_score = (const float*)d_in[0];
  const float* node_repr  = (const float*)d_in[1];
  const float* rel_emb    = (const float*)d_in[2];
  const float* qse        = (const float*)d_in[3];
  const float* qre        = (const float*)d_in[4];
  const float* Wq         = (const float*)d_in[5];
  const float* Wk         = (const float*)d_in[6];
  const float* W_lin      = (const float*)d_in[7];
  const float* b_lin      = (const float*)d_in[8];
  const int*   edges      = (const int*)d_in[9];
  (void)in_sizes; (void)n_in; (void)out_size; (void)ws_size;

  float* outscore = (float*)d_out;          // updated_node_score [NN]
  float* upd      = (float*)d_out + NN;     // out_repr region doubles as agg scratch

  char* w = (char*)d_ws;                    // ~5 MB total, 256B-aligned sections
  float*    Mfull  = (float*)(w + 0);       // 256*256*4      = 262144
  uint16_t* Abf    = (uint16_t*)(w + 262144);   // 256*128*2  =  65536
  float*    gq     = (float*)(w + 327680);      // 128*256*4  = 131072
  float*    logits = (float*)(w + 458752);      // NE*4       = 524288
  float*    exbuf  = (float*)(w + 983040);      // NE*4
  float*    softb  = (float*)(w + 1507328);     // NE*4
  float*    tscore = (float*)(w + 2031616);     // NE*4
  uint32_t* segmax = (uint32_t*)(w + 2555904);  // NN*4       = 800000
  float*    segsum = (float*)(w + 3355904);     // NN*4
  int*      flags  = (int*)(w + 4155904);       // NN*4
  int*      sel    = (int*)(w + 4955904);       // PSEL*4     =  65536

  k_buildM  <<<256, 256, 0, stream>>>(Wq, Wk, Mfull, Abf);
  k_gq      <<<QN, 256, 0, stream>>>(Mfull, qse, qre, gq);
  k_fill    <<<(NN + 255)/256, 256, 0, stream>>>(segmax, segsum, flags, outscore);
  k_edge    <<<NEE/16, 128, 0, stream>>>(node_repr, rel_emb, qse, qre, edges, Abf, gq, logits);
  k_segmax  <<<NEE/256, 256, 0, stream>>>(logits, edges, segmax);
  k_expsum  <<<NEE/256, 256, 0, stream>>>(logits, edges, segmax, exbuf, segsum);
  k_soft    <<<NEE/256, 256, 0, stream>>>(exbuf, edges, segsum, node_score, softb, tscore);
  k_topk    <<<QN, 512, 0, stream>>>(tscore, sel);
  k_scatter1<<<PSEL/256, 256, 0, stream>>>(sel, edges, softb, node_score, outscore, flags);
  k_initupd <<<NN*16/256, 256, 0, stream>>>(node_repr, flags, upd);
  k_scatter2<<<PSEL, 64, 0, stream>>>(sel, edges, softb, node_repr, upd);
  k_out     <<<NN/64, 128, 0, stream>>>(upd, W_lin, b_lin);
}
